// GraphConv_29145648070885
// MI455X (gfx1250) — matrix operand order repaired and checked by reading, not compiled, not run on hardware
//
#include <hip/hip_runtime.h>
#include <hip/hip_bf16.h>

typedef __attribute__((ext_vector_type(16))) __bf16 v16bf;
typedef __attribute__((ext_vector_type(8)))  float  v8f;

#define B_   8
#define N_   2048
#define F_   128
#define O_   128
#define EPS_ 1e-6f

// ---------------------------------------------------------------------------
// CDNA5 async-copy helpers (gfx1250): global -> LDS DMA, tracked by ASYNCcnt
// ---------------------------------------------------------------------------
__device__ __forceinline__ void async_g2l_b128(const void* gptr, void* lptr) {
    // vdst = 32-bit LDS byte address, vaddr = 64-bit global address (GV mode)
    asm volatile("global_load_async_to_lds_b128 %0, %1, off"
                 :: "v"((unsigned)(unsigned long long)lptr),
                    "v"((unsigned long long)gptr)
                 : "memory");
}
__device__ __forceinline__ void wait_async0() {
    asm volatile("s_wait_asynccnt 0x0" ::: "memory");
}

// ---------------------------------------------------------------------------
// Kernel 1: d[b,n] = 1 / (eps + sqrt( sum_{m!=n} adj[b,n,m] + 1 ))
// ---------------------------------------------------------------------------
__global__ void gc_degree(const float* __restrict__ adj, float* __restrict__ d) {
    const int n = blockIdx.x, b = blockIdx.y, t = threadIdx.x;
    const float* row = adj + ((size_t)b * N_ + n) * (size_t)N_;
    float s = 0.0f;
    #pragma unroll
    for (int i = 0; i < N_ / 256; ++i) s += row[t + 256 * i];
    __shared__ float red[256];
    red[t] = s;
    __syncthreads();
    for (int w = 128; w > 0; w >>= 1) {
        if (t < w) red[t] += red[t + w];
        __syncthreads();
    }
    if (t == 0) {
        float total = red[0] - row[n] + 1.0f;   // zero diagonal, add self-loop
        d[b * N_ + n] = 1.0f / (EPS_ + sqrtf(total));
    }
}

// ---------------------------------------------------------------------------
// Kernel 2: xsT[b][f][m] = bf16( d[b,m] * x[b,m,f] )   (m contiguous)
// ---------------------------------------------------------------------------
__global__ void gc_prep_x(const float* __restrict__ x, const float* __restrict__ d,
                          __bf16* __restrict__ xsT) {
    const int idx = blockIdx.x * 256 + threadIdx.x;
    const int m = idx % N_;
    const int f = (idx / N_) % F_;
    const int b = idx / (N_ * F_);
    xsT[idx] = (__bf16)(x[((size_t)b * N_ + m) * F_ + f] * d[b * N_ + m]);
}

// ---------------------------------------------------------------------------
// Kernel 3: wT[o][f] = bf16( W[f][o] )   (f contiguous)
// ---------------------------------------------------------------------------
__global__ void gc_prep_w(const float* __restrict__ W, __bf16* __restrict__ wT) {
    const int idx = blockIdx.x * 256 + threadIdx.x;
    const int f = idx % F_;
    const int o = idx / F_;
    wT[idx] = (__bf16)W[f * O_ + o];
}

// ---------------------------------------------------------------------------
// A-tile staging helpers (register-pipelined, bf16 convert + diagonal fix)
// ---------------------------------------------------------------------------
__device__ __forceinline__ void load_a_regs(float4* areg, const float* __restrict__ adj,
                                            int b, int n0, int kk, int t) {
    #pragma unroll
    for (int k2 = 0; k2 < 4; ++k2) {
        const int i  = t + 256 * k2;
        const int r  = i >> 4;
        const int c4 = (i & 15) << 2;
        areg[k2] = *(const float4*)(adj + ((size_t)b * N_ + n0 + r) * (size_t)N_ + kk + c4);
    }
}
__device__ __forceinline__ void store_a_tile(const float4* areg, __bf16* aDst,
                                             int n0, int kk, int t) {
    #pragma unroll
    for (int k2 = 0; k2 < 4; ++k2) {
        const int i  = t + 256 * k2;
        const int r  = i >> 4;
        const int c4 = (i & 15) << 2;
        const int grow = n0 + r;
        const float vv[4] = {areg[k2].x, areg[k2].y, areg[k2].z, areg[k2].w};
        __bf16* dst = aDst + r * 64 + c4;
        #pragma unroll
        for (int j = 0; j < 4; ++j) {
            const int gcol = kk + c4 + j;
            dst[j] = (__bf16)((gcol == grow) ? 1.0f : vv[j]);
        }
    }
}
__device__ __forceinline__ void async_b_tile(const __bf16* __restrict__ xb_kk,
                                             __bf16* bDst, int t) {
    #pragma unroll
    for (int k2 = 0; k2 < 4; ++k2) {        // 4 x b128 per lane = 16KB/block
        const int i = t + 256 * k2;
        const int f = i >> 3;
        const int c = (i & 7) << 3;
        async_g2l_b128(xb_kk + (size_t)f * N_ + c, bDst + f * 64 + c);
    }
}

// ---------------------------------------------------------------------------
// Kernel 4: fused  out = relu( diag(d) * A_hat * xs @ W )
// 256 thr (8 wave32), tile 64 x 128, double-buffered LDS, async B staging
// ---------------------------------------------------------------------------
__global__ void __launch_bounds__(256)
gc_main(const float* __restrict__ adj, const __bf16* __restrict__ xsT,
        const __bf16* __restrict__ wT, const float* __restrict__ ddeg,
        float* __restrict__ out) {
    const int t      = threadIdx.x;
    const int lane   = t & 31;
    const int wv     = t >> 5;
    const int lane16 = lane & 15;
    const int hi     = lane >> 4;
    const int base0  = hi * 16;          // A-frag K sub-offset
    const int kb16   = hi * 16;          // B-frag K sub-offset
    const int ms = (wv & 3) * 16;        // row slab
    const int fh = (wv >> 2) * 64;       // F/O half

    const int n0 = blockIdx.x * 64;
    const int b  = blockIdx.y;

    // LDS: [0,8K) aT0, [8K,16K) aT1, [16K,32K) bT0, [32K,48K) bT1
    // phase 2: [0,16K) cT, [16K,48K) wL
    __shared__ __align__(16) unsigned char lds[49152];
    __bf16* cT = (__bf16*)lds;             // [64][128]  (phase 2, reuses aT)
    __bf16* wL = (__bf16*)(lds + 16384);   // [128][128] (phase 2, reuses bT)

    const __bf16* xb = xsT + (size_t)b * F_ * (size_t)N_;

    // ---- prologue: stage tile 0
    float4 areg[4];
    async_b_tile(xb, (__bf16*)(lds + 16384), t);      // ASYNCcnt += 4
    load_a_regs(areg, adj, b, n0, 0, t);
    store_a_tile(areg, (__bf16*)lds, n0, 0, t);

    v8f acc[4] = {};

    for (int it = 0; it < 32; ++it) {
        const int kk  = it * 64;
        const int cur = it & 1;
        const int nxt = cur ^ 1;
        __bf16* aCur = (__bf16*)(lds + cur * 8192);
        __bf16* aNxt = (__bf16*)(lds + nxt * 8192);
        __bf16* bCur = (__bf16*)(lds + 16384 + cur * 16384);
        __bf16* bNxt = (__bf16*)(lds + 16384 + nxt * 16384);

        wait_async0();        // own B loads for tile `it` landed in LDS
        __syncthreads();      // all waves: tile `it` staged, tile `it-1` reads done

        const bool more = (it + 1 < 32);
        if (more) {
            async_b_tile(xb + kk + 64, bNxt, t);      // overlaps with WMMAs below
            load_a_regs(areg, adj, b, n0, kk + 64, t);
            if (it + 2 < 32)                          // L2 prefetch 2 tiles ahead
                __builtin_prefetch(adj + ((size_t)b * N_ + n0 + (t & 63)) * (size_t)N_
                                       + kk + 128, 0, 0);
        }

        // ---- 8 WMMAs per wave (2 K-substeps x 4 F-tiles)
        #pragma unroll
        for (int ks = 0; ks < 64; ks += 32) {
            union { uint4 u[2]; v16bf v; } af;
            const __bf16* ar = aCur + (ms + lane16) * 64 + ks;
            af.u[0] = *(const uint4*)(ar + base0);
            af.u[1] = *(const uint4*)(ar + 8 + base0);
            #pragma unroll
            for (int ft = 0; ft < 4; ++ft) {
                union { uint4 u[2]; v16bf v; } bfrag;
                const __bf16* br = bCur + (fh + ft * 16 + lane16) * 64 + ks + kb16;
                bfrag.u[0] = ((const uint4*)br)[0];
                bfrag.u[1] = ((const uint4*)br)[1];
                acc[ft] = __builtin_amdgcn_wmma_f32_16x16x32_bf16(
                    false, af.v, false, bfrag.v, (short)0, acc[ft], false, false);
            }
        }

        if (more) store_a_tile(areg, aNxt, n0, kk + 64, t);
    }
    __syncthreads();   // all K-loop LDS reads retired; buffers reusable

    // ---- stage W^T into LDS via async DMA (overlaps with cT stores below)
    #pragma unroll
    for (int k2 = 0; k2 < 8; ++k2) {       // 2048 x b128 total
        const int i = t + 256 * k2;
        async_g2l_b128(wT + i * 8, wL + i * 8);
    }

    // ---- scale rows by d_n, write agg tile (bf16) into cT
    float dsc[8];
    #pragma unroll
    for (int v = 0; v < 8; ++v)
        dsc[v] = ddeg[b * N_ + n0 + ms + v + hi * 8];
    #pragma unroll
    for (int ft = 0; ft < 4; ++ft)
        #pragma unroll
        for (int v = 0; v < 8; ++v)
            cT[(ms + v + hi * 8) * 128 + fh + ft * 16 + lane16] =
                (__bf16)(acc[ft][v] * dsc[v]);

    wait_async0();
    __syncthreads();

    // ---- second GEMM: out_tile = agg_tile @ W   (K=128, 16 WMMAs per wave)
    v8f acc2[4] = {};
    #pragma unroll
    for (int ks = 0; ks < 128; ks += 32) {
        union { uint4 u[2]; v16bf v; } af;
        const __bf16* ar = cT + (ms + lane16) * 128 + ks;
        af.u[0] = *(const uint4*)(ar + base0);
        af.u[1] = *(const uint4*)(ar + 8 + base0);
        #pragma unroll
        for (int ot = 0; ot < 4; ++ot) {
            union { uint4 u[2]; v16bf v; } bfrag;
            const __bf16* br = wL + (fh + ot * 16 + lane16) * 128 + ks + kb16;
            bfrag.u[0] = ((const uint4*)br)[0];
            bfrag.u[1] = ((const uint4*)br)[1];
            acc2[ot] = __builtin_amdgcn_wmma_f32_16x16x32_bf16(
                false, af.v, false, bfrag.v, (short)0, acc2[ot], false, false);
        }
    }

    // ---- ReLU + fp32 store
    #pragma unroll
    for (int ot = 0; ot < 4; ++ot)
        #pragma unroll
        for (int v = 0; v < 8; ++v) {
            float r = acc2[ot][v];
            r = r > 0.0f ? r : 0.0f;
            out[((size_t)b * N_ + n0 + ms + v + hi * 8) * O_ + fh + ot * 16 + lane16] = r;
        }
}

// ---------------------------------------------------------------------------
extern "C" void kernel_launch(void* const* d_in, const int* in_sizes, int n_in,
                              void* d_out, int out_size, void* d_ws, size_t ws_size,
                              hipStream_t stream) {
    const float* x   = (const float*)d_in[0];   // [B,N,F]
    const float* adj = (const float*)d_in[1];   // [B,N,N]
    const float* W   = (const float*)d_in[2];   // [F,O]
    float* out = (float*)d_out;                 // [B,N,O] fp32

    float*  d_deg = (float*)d_ws;                                   // 64 KB
    __bf16* xsT   = (__bf16*)((char*)d_ws + 65536);                 // 4 MB
    __bf16* wT    = (__bf16*)((char*)d_ws + 65536 + 4194304);       // 32 KB

    gc_degree<<<dim3(N_, B_), 256, 0, stream>>>(adj, d_deg);
    gc_prep_x<<<(B_ * F_ * N_) / 256, 256, 0, stream>>>(x, d_deg, xsT);
    gc_prep_w<<<(O_ * F_) / 256, 256, 0, stream>>>(W, wT);
    gc_main<<<dim3(N_ / 64, B_), 256, 0, stream>>>(adj, xsT, wT, d_deg, out);
}
